// PlasDyn_18597208391865
// MI455X (gfx1250) — matrix operand order repaired and checked by reading, not compiled, not run on hardware
//
#include <hip/hip_runtime.h>
#include <hip/hip_bf16.h>

// Problem sizes (fixed by the reference).
static constexpr int    Hd  = 1024;
static constexpr int    Bd  = 64;
static constexpr int    Td  = 512;
static constexpr size_t HHd = (size_t)Hd * Hd;

typedef _Float16 v16h __attribute__((ext_vector_type(16)));
typedef _Float16 v8h  __attribute__((ext_vector_type(8)));
typedef float    v8f  __attribute__((ext_vector_type(8)));

// ---------- helpers ----------

// Build a 16-element f16 fragment from two 16-byte contiguous runs.
__device__ __forceinline__ v16h ld16(const _Float16* p0, const _Float16* p1) {
    v8h lo = *(const v8h*)p0;
    v8h hi = *(const v8h*)p1;
    return __builtin_shufflevector(lo, hi, 0,1,2,3,4,5,6,7,8,9,10,11,12,13,14,15);
}

__device__ __forceinline__ v8f wmma_f16(v16h a, v16h b, v8f c) {
    // D = A(16x32 f16) * B(32x16 f16) + C(16x16 f32)
    return __builtin_amdgcn_wmma_f32_16x16x32_f16(
        /*neg_a=*/false, a, /*neg_b=*/false, b,
        /*c_mod=*/(short)0, c, /*reuse_a=*/false, /*reuse_b=*/false);
}

__device__ __forceinline__ float sigmoidf_(float x) {
    return 1.0f / (1.0f + __expf(-x));
}

// Grid-wide split barrier: count + generation, s_sleep spin.
__device__ __forceinline__ void grid_barrier(unsigned* cnt, unsigned* gen, unsigned nwg) {
    __syncthreads();
    if (threadIdx.x == 0) {
        unsigned g = __hip_atomic_load(gen, __ATOMIC_ACQUIRE, __HIP_MEMORY_SCOPE_AGENT);
        __threadfence();
        unsigned prev = __hip_atomic_fetch_add(cnt, 1u, __ATOMIC_ACQ_REL, __HIP_MEMORY_SCOPE_AGENT);
        if (prev == nwg - 1u) {
            __hip_atomic_store(cnt, 0u, __ATOMIC_RELAXED, __HIP_MEMORY_SCOPE_AGENT);
            __hip_atomic_fetch_add(gen, 1u, __ATOMIC_ACQ_REL, __HIP_MEMORY_SCOPE_AGENT);
        } else {
            while (__hip_atomic_load(gen, __ATOMIC_ACQUIRE, __HIP_MEMORY_SCOPE_AGENT) == g) {
                __builtin_amdgcn_s_sleep(2);
            }
        }
        __threadfence();
    }
    __syncthreads();
}

// ---------- kernels ----------

// fp32 -> fp16 conversion of x [B,T,H]
__global__ __launch_bounds__(256) void cvt_x_kernel(const float* __restrict__ x,
                                                    _Float16* __restrict__ xh, int n) {
    int stride = (int)(gridDim.x * blockDim.x);
    for (int i = blockIdx.x * blockDim.x + threadIdx.x; i < n; i += stride)
        xh[i] = (_Float16)x[i];
}

// Pack 4 weight matrices into fp16 buffer [Wmu | Wgmu | Wa | Wga], row-major (H,H).
__global__ __launch_bounds__(256) void cvt_w_kernel(const float* __restrict__ Wmu,
                                                    const float* __restrict__ Wgmu,
                                                    const float* __restrict__ Wa,
                                                    const float* __restrict__ Wga,
                                                    _Float16* __restrict__ W16) {
    int n = (int)(4 * HHd);
    int stride = (int)(gridDim.x * blockDim.x);
    for (int i = blockIdx.x * blockDim.x + threadIdx.x; i < n; i += stride) {
        int sel = i >> 20;                  // HHd == 2^20
        int off = i & ((1 << 20) - 1);
        const float* src = (sel == 0) ? Wmu : (sel == 1) ? Wgmu : (sel == 2) ? Wa : Wga;
        W16[i] = (_Float16)src[off];
    }
}

__global__ void init_bar_kernel(unsigned* bar) { bar[0] = 0u; bar[1] = 0u; }

// Precompute Ax[t,b,:] = x[b,t,:] @ W_a^T + b_a   (big parallel WMMA GEMM).
// Tile id = global wave id; mt in [0,2048) covers (t, b-block), nt in [0,64).
__global__ __launch_bounds__(256) void ax_gemm_kernel(const _Float16* __restrict__ xh,
                                                      const _Float16* __restrict__ Wa16,
                                                      const float* __restrict__ b_a,
                                                      float* __restrict__ Ax) {
    int wave = threadIdx.x >> 5;
    int lane = threadIdx.x & 31;
    int tid  = blockIdx.x * 8 + wave;        // 0..131071
    int mt   = tid >> 6;                     // 0..2047
    int nt   = tid & 63;                     // 0..63
    int t    = mt >> 2;
    int b0   = (mt & 3) << 4;
    int half = lane >> 4;
    int l16  = lane & 15;

    const _Float16* arow = xh + ((size_t)(b0 + l16) * Td + t) * Hd;  // x[b,t,:]
    int ncol = nt * 16 + l16;
    const _Float16* brow = Wa16 + (size_t)ncol * Hd;                 // W_a row n == B column n

    v8f acc = {};
    for (int k = 0; k < Hd; k += 32) {
        v16h a = ld16(arow + k + 16 * half, arow + k + 16 * half + 8);
        v16h b = ld16(brow + k + 16 * half, brow + k + 16 * half + 8);
        acc = wmma_f16(a, b, acc);
    }
    float bias = b_a[ncol];
#pragma unroll
    for (int i = 0; i < 8; ++i) {
        int b = b0 + i + 8 * half;                       // C layout: M = i + 8*(lane/16)
        Ax[((size_t)t * Bd + b) * Hd + ncol] = acc[i] + bias;
    }
}

// Persistent recurrence: 32 WGs x 8 waves = 256 waves, one 16x16 h-tile each.
// Per step: 3 WMMA GEMMs (W_mu, Wg_mu, Wg_a) over K=1024, fused gate epilogue,
// double-buffered fp16 h, grid barrier between steps.
__global__ __launch_bounds__(256) void recur_kernel(const _Float16* __restrict__ W16,
                                                    const float* __restrict__ b_mu,
                                                    const float* __restrict__ bg_mu,
                                                    const float* __restrict__ bg_a,
                                                    const float* __restrict__ Ax,
                                                    _Float16* __restrict__ h16,  // [2][B*H]
                                                    float* __restrict__ out,     // [B*H]
                                                    unsigned* __restrict__ bar) {
    constexpr unsigned NWG = 32;
    int wave = threadIdx.x >> 5;
    int lane = threadIdx.x & 31;
    int tid  = blockIdx.x * 8 + wave;    // 0..255
    int mt   = tid >> 6;                 // 0..3  (rows of B dim)
    int nt   = tid & 63;                 // 0..63 (cols of H dim)
    int half = lane >> 4;
    int l16  = lane & 15;
    int ncol = nt * 16 + l16;

    const _Float16* brow_mu  = W16 + 0 * HHd + (size_t)ncol * Hd;
    const _Float16* brow_gmu = W16 + 1 * HHd + (size_t)ncol * Hd;
    const _Float16* brow_ga  = W16 + 3 * HHd + (size_t)ncol * Hd;
    float bmu  = b_mu[ncol];
    float bgmu = bg_mu[ncol];
    float bga  = bg_a[ncol];

    // h0 = 0: each wave zeroes its own tile in buffer 0.
#pragma unroll
    for (int i = 0; i < 8; ++i) {
        int r = mt * 16 + i + 8 * half;
        h16[(size_t)r * Hd + ncol] = (_Float16)0.0f;
    }
    grid_barrier(&bar[0], &bar[1], NWG);

    for (int t = 0; t < Td; ++t) {
        const _Float16* hcur  = h16 + (size_t)(t & 1) * (Bd * Hd);
        _Float16*       hnext = h16 + (size_t)((t + 1) & 1) * (Bd * Hd);
        const _Float16* arow  = hcur + (size_t)(mt * 16 + l16) * Hd;

        // Prefetch next step's Ax tile into cache while we crunch WMMAs.
        if (t + 1 < Td) {
            const float* axn = Ax + ((size_t)(t + 1) * Bd + mt * 16) * Hd + ncol;
#pragma unroll
            for (int i = 0; i < 8; ++i)
                __builtin_prefetch(axn + (size_t)(i + 8 * half) * Hd, 0, 1);
        }

        v8f amu = {}, agmu = {}, aga = {};
        for (int k = 0; k < Hd; k += 32) {
            v16h a  = ld16(arow + k + 16 * half, arow + k + 16 * half + 8);
            v16h b0 = ld16(brow_mu  + k + 16 * half, brow_mu  + k + 16 * half + 8);
            amu  = wmma_f16(a, b0, amu);
            v16h b1 = ld16(brow_gmu + k + 16 * half, brow_gmu + k + 16 * half + 8);
            agmu = wmma_f16(a, b1, agmu);
            v16h b2 = ld16(brow_ga  + k + 16 * half, brow_ga  + k + 16 * half + 8);
            aga  = wmma_f16(a, b2, aga);
        }

        const float* axp = Ax + ((size_t)t * Bd + mt * 16) * Hd;
#pragma unroll
        for (int i = 0; i < 8; ++i) {
            int rl = i + 8 * half;                         // local row in tile
            float mu = amu[i] + bmu;
            float g1 = agmu[i] + bgmu;
            float g2 = aga[i] + bga;
            float ax = axp[(size_t)rl * Hd + ncol];
            float hv = mu * sigmoidf_(g1) + ax * sigmoidf_(g2);
            hnext[(size_t)(mt * 16 + rl) * Hd + ncol] = (_Float16)hv;
            if (t == Td - 1)
                out[(size_t)(mt * 16 + rl) * Hd + ncol] = hv;
        }
        grid_barrier(&bar[0], &bar[1], NWG);
    }
}

// ---------- launch ----------

extern "C" void kernel_launch(void* const* d_in, const int* in_sizes, int n_in,
                              void* d_out, int out_size, void* d_ws, size_t ws_size,
                              hipStream_t stream) {
    const float* x     = (const float*)d_in[0];
    const float* W_mu  = (const float*)d_in[1];
    const float* b_mu  = (const float*)d_in[2];
    const float* Wg_mu = (const float*)d_in[3];
    const float* bg_mu = (const float*)d_in[4];
    const float* W_a   = (const float*)d_in[5];
    const float* b_a   = (const float*)d_in[6];
    const float* Wg_a  = (const float*)d_in[7];
    const float* bg_a  = (const float*)d_in[8];

    // Workspace layout (all offsets 256-B aligned):
    //   Ax  fp32 [T,B,H]   @ 0          (134,217,728 B)
    //   xh  fp16 [B,T,H]   @ 134217728  ( 67,108,864 B)
    //   W16 fp16 [4][H,H]  @ 201326592  (  8,388,608 B)
    //   h16 fp16 [2][B,H]  @ 209715200  (    262,144 B)
    //   bar u32 [2]        @ 209977344
    char* ws = (char*)d_ws;
    float*    Ax  = (float*)(ws);
    _Float16* xh  = (_Float16*)(ws + 134217728u);
    _Float16* W16 = (_Float16*)(ws + 201326592u);
    _Float16* h16 = (_Float16*)(ws + 209715200u);
    unsigned* bar = (unsigned*)(ws + 209977344u);

    cvt_x_kernel<<<4096, 256, 0, stream>>>(x, xh, Bd * Td * Hd);
    cvt_w_kernel<<<2048, 256, 0, stream>>>(W_mu, Wg_mu, W_a, Wg_a, W16);
    init_bar_kernel<<<1, 1, 0, stream>>>(bar);

    // Parallel precompute of the input-dependent gate term (one wave per 16x16 tile).
    ax_gemm_kernel<<<16384, 256, 0, stream>>>(xh, W16 + 2 * HHd, b_a, Ax);

    // Persistent sequential recurrence.
    recur_kernel<<<32, 256, 0, stream>>>(W16, b_mu, bg_mu, bg_a, Ax, h16,
                                         (float*)d_out, bar);
}
